// SelfAttention_20753281975098
// MI455X (gfx1250) — matrix operand order repaired and checked by reading, not compiled, not run on hardware
//
#include <hip/hip_runtime.h>

// Problem constants (match the reference)
#define B_ 8
#define S_ 4096
#define D_ 128   // model dim == head dim == 128

typedef __attribute__((ext_vector_type(16))) __bf16 v16bf;
typedef __attribute__((ext_vector_type(8)))  __bf16 v8bf;
typedef __attribute__((ext_vector_type(8)))  float  v8f;
typedef __attribute__((ext_vector_type(4)))  float  v4f;

__device__ __forceinline__ v8f wmma_bf16(v16bf a, v16bf b, v8f c) {
  // D = A(16x32 bf16) * B(32x16 bf16) + C(16x16 f32)
  return __builtin_amdgcn_wmma_f32_16x16x32_bf16(
      /*neg_a=*/false, a, /*neg_b=*/false, b,
      /*c_mod=*/(short)0, c, /*reuse_a=*/false, /*reuse_b=*/false);
}

// Butterfly reductions over a 16-lane half of the wave (masks 1,2,4,8 never
// cross the lane-16 boundary, matching the C-matrix row grouping).
__device__ __forceinline__ float half_max(float t) {
  t = fmaxf(t, __shfl_xor(t, 1, 32));
  t = fmaxf(t, __shfl_xor(t, 2, 32));
  t = fmaxf(t, __shfl_xor(t, 4, 32));
  t = fmaxf(t, __shfl_xor(t, 8, 32));
  return t;
}
__device__ __forceinline__ float half_sum(float t) {
  t += __shfl_xor(t, 1, 32);
  t += __shfl_xor(t, 2, 32);
  t += __shfl_xor(t, 4, 32);
  t += __shfl_xor(t, 8, 32);
  return t;
}

// ---------------------------------------------------------------------------
// Pass 0: convert the three 128x128 f32 weight matrices into TRANSPOSED bf16
// (Wt[m][col][k]) so pass-1 B operands are single contiguous 32B loads.
// Trivial cost: 3*128*128 elements total.
// ---------------------------------------------------------------------------
__global__ __launch_bounds__(128) void prep_weights(
    const float* __restrict__ Wq, const float* __restrict__ Wk,
    const float* __restrict__ Wv, __bf16* __restrict__ Wt) {
  const int col = blockIdx.x;   // 0..127
  const int m   = blockIdx.y;   // 0..2
  const int k   = threadIdx.x;  // 0..127
  const float* W = (m == 0) ? Wq : (m == 1) ? Wk : Wv;
  Wt[((size_t)m * D_ + col) * D_ + k] = (__bf16)W[(size_t)k * D_ + col];
}

// ---------------------------------------------------------------------------
// Pass 1: xp = x + PE;  Q = xp*Wq+bq ; K = xp*Wk+bk ; V = xp*Wv+bv
// Q,K stored row-major bf16 [B,S,128]; V stored transposed bf16 [B,128,S].
// One wave handles a 16-row tile; 12 WMMAs per col-tile iteration group.
// ---------------------------------------------------------------------------
__global__ __launch_bounds__(32) void qkv_pe_proj(
    const float* __restrict__ x, const __bf16* __restrict__ Wt,
    const float* __restrict__ bq, const float* __restrict__ bk,
    const float* __restrict__ bv, __bf16* __restrict__ Qbf,
    __bf16* __restrict__ Kbf, __bf16* __restrict__ Vt) {
  const int l  = threadIdx.x;      // lane 0..31 (wave32)
  const int r  = l & 15;           // A-operand row = lane%16
  const int h  = l >> 4;           // lane half
  const int s0 = blockIdx.x * 16;  // sequence tile
  const int b  = blockIdx.y;       // batch
  const int s  = s0 + r;

  const float* xrow = x + ((size_t)b * S_ + s) * D_;
  const float  sf   = (float)s;

  // Build the four 16x32 A-operand chunks of (x + PE) in bf16.
  // A k order used here (same as B): lane half h -> K in {16h..16h+15}
  v16bf a[4];
  const float kPE = 0.07195578411f;  // ln(10000)/128
#pragma unroll
  for (int dc = 0; dc < 4; ++dc) {
#pragma unroll
    for (int i = 0; i < 8; ++i) {
      int k = dc * 32 + 16 * h + i;
      float div = __expf(-(float)(k & ~1) * kPE);
      float pe  = (k & 1) ? __cosf(sf * div) : __sinf(sf * div);
      a[dc][i] = (__bf16)(xrow[k] + pe);
    }
#pragma unroll
    for (int i = 0; i < 8; ++i) {
      int k = dc * 32 + 16 * h + 8 + i;
      float div = __expf(-(float)(k & ~1) * kPE);
      float pe  = (k & 1) ? __cosf(sf * div) : __sinf(sf * div);
      a[dc][8 + i] = (__bf16)(xrow[k] + pe);
    }
  }

  const float* bias[3] = {bq, bk, bv};

#pragma unroll
  for (int m3 = 0; m3 < 3; ++m3) {
    const __bf16* WtM = Wt + (size_t)m3 * D_ * D_;
    for (int t = 0; t < 8; ++t) {  // 8 col-tiles of 16 over A=128
      const int col = t * 16 + r;  // B-operand N = lane%16
      float bb0 = bias[m3][col];
      v8f c;
#pragma unroll
      for (int v = 0; v < 8; ++v) c[v] = bb0;  // bias-initialized accumulator
#pragma unroll
      for (int dc = 0; dc < 4; ++dc) {
        // Pre-transposed bf16 weights: contiguous 32B load per lane.
        // B k order used here (same as A): lanes 0-15 K=0..15, lanes 16-31 K=16..31
        v16bf bb =
            *(const v16bf*)(WtM + (size_t)col * D_ + dc * 32 + 16 * h);
        c = wmma_bf16(a[dc], bb, c);
      }
      // C layout: VGPR v, lane half h -> row v + 8h, col = lane%16
#pragma unroll
      for (int v = 0; v < 8; ++v) {
        int row = v + 8 * h;
        __bf16 val = (__bf16)c[v];
        if (m3 == 0)
          Qbf[((size_t)b * S_ + s0 + row) * D_ + col] = val;
        else if (m3 == 1)
          Kbf[((size_t)b * S_ + s0 + row) * D_ + col] = val;
        else
          Vt[((size_t)b * D_ + col) * S_ + s0 + row] = val;  // transposed V
      }
    }
  }
}

// ---------------------------------------------------------------------------
// Pass 2: flash attention. One wave per (batch, 16-query tile).
// Key loop step = 32 so the PV product is a single K=32 WMMA per col-tile.
// ---------------------------------------------------------------------------
__global__ __launch_bounds__(32) void flash_attn(
    const __bf16* __restrict__ Qbf, const __bf16* __restrict__ Kbf,
    const __bf16* __restrict__ Vt, float* __restrict__ out) {
  // P-tile staging. Row stride 36 floats = 144B: 16B-aligned rows for
  // vector ds reads; 36r mod 64 banks is conflict-free across 16 lanes.
  __shared__ __align__(16) float psm[16][36];

  const int l  = threadIdx.x;
  const int r  = l & 15;
  const int h  = l >> 4;
  const int s0 = blockIdx.x * 16;
  const int b  = blockIdx.y;

  const __bf16* Qb = Qbf + (size_t)b * S_ * D_;
  const __bf16* Kb = Kbf + (size_t)b * S_ * D_;
  const __bf16* Vb = Vt + (size_t)b * D_ * S_;

  // Load the 16x128 Q tile once, already in A-operand layout.
  v16bf qa[4];
#pragma unroll
  for (int dc = 0; dc < 4; ++dc) {
    const v8bf lo = *(const v8bf*)(Qb + (size_t)(s0 + r) * D_ + dc * 32 + 16 * h);
    const v8bf hi =
        *(const v8bf*)(Qb + (size_t)(s0 + r) * D_ + dc * 32 + 16 * h + 8);
    qa[dc] = __builtin_shufflevector(lo, hi, 0, 1, 2, 3, 4, 5, 6, 7, 8, 9, 10,
                                     11, 12, 13, 14, 15);
  }

  v8f o[8];
#pragma unroll
  for (int t = 0; t < 8; ++t)
#pragma unroll
    for (int v = 0; v < 8; ++v) o[t][v] = 0.0f;

  float mrow[8], lrow[8];
#pragma unroll
  for (int v = 0; v < 8; ++v) { mrow[v] = -1e30f; lrow[v] = 0.0f; }

  const float scale = 0.08838834764831845f;  // 1/sqrt(128)

  for (int k0 = 0; k0 < S_; k0 += 32) {
    // Prefetch next key tile (global_prefetch_b8) while this one computes.
    if (k0 + 32 < S_) {
      __builtin_prefetch(Kb + (size_t)(k0 + 32 + r) * D_, 0, 3);
      __builtin_prefetch(Kb + (size_t)(k0 + 48 + r) * D_, 0, 3);
    }

    // ---- scores for 32 keys: two 16x16 C tiles, 4 d-chunks each ----
    v8f c0, c1;
#pragma unroll
    for (int v = 0; v < 8; ++v) { c0[v] = 0.0f; c1[v] = 0.0f; }
#pragma unroll
    for (int dc = 0; dc < 4; ++dc) {
      // B operand = K^T chunk: lane N selects key row -> contiguous 32B load
      v16bf bk0 = *(const v16bf*)(Kb + (size_t)(k0 + r) * D_ + dc * 32 + 16 * h);
      c0 = wmma_bf16(qa[dc], bk0, c0);
    }
#pragma unroll
    for (int dc = 0; dc < 4; ++dc) {
      v16bf bk1 =
          *(const v16bf*)(Kb + (size_t)(k0 + 16 + r) * D_ + dc * 32 + 16 * h);
      c1 = wmma_bf16(qa[dc], bk1, c1);
    }

    // ---- online softmax over the 32 new keys, per accumulator row ----
#pragma unroll
    for (int v = 0; v < 8; ++v) {
      float sv0 = c0[v] * scale;
      float sv1 = c1[v] * scale;
      float tmax = half_max(fmaxf(sv0, sv1));
      float newm = fmaxf(mrow[v], tmax);
      float p0 = __expf(sv0 - newm);
      float p1 = __expf(sv1 - newm);
      float corr = __expf(mrow[v] - newm);
      lrow[v] = lrow[v] * corr + half_sum(p0 + p1);
      mrow[v] = newm;
#pragma unroll
      for (int t = 0; t < 8; ++t) o[t][v] *= corr;
      int row = v + 8 * h;        // C-layout row
      psm[row][r] = p0;           // keys k0 .. k0+15
      psm[row][16 + r] = p1;      // keys k0+16 .. k0+31
    }
    __syncthreads();

    // ---- re-read P in A-operand layout via aligned float4 ds loads ----
    v16bf pa;
    {
      const v4f p0v = *(const v4f*)&psm[r][16 * h];
      const v4f p1v = *(const v4f*)&psm[r][16 * h + 4];
      const v4f p2v = *(const v4f*)&psm[r][16 * h + 8];
      const v4f p3v = *(const v4f*)&psm[r][16 * h + 12];
#pragma unroll
      for (int i = 0; i < 4; ++i) {
        pa[i]      = (__bf16)p0v[i];
        pa[4 + i]  = (__bf16)p1v[i];
        pa[8 + i]  = (__bf16)p2v[i];
        pa[12 + i] = (__bf16)p3v[i];
      }
    }

    // ---- out += P (16x32) * V (32x16) per col-tile ----
#pragma unroll
    for (int t = 0; t < 8; ++t) {
      // V stored transposed -> contiguous 32B load along keys per lane
      v16bf bv =
          *(const v16bf*)(Vb + (size_t)(t * 16 + r) * S_ + k0 + 16 * h);
      if (k0 + 32 < S_)
        __builtin_prefetch(Vb + (size_t)(t * 16 + r) * S_ + k0 + 32, 0, 3);
      o[t] = wmma_bf16(pa, bv, o[t]);
    }
    __syncthreads();
  }

  // ---- epilogue: divide by softmax denominator, store f32 ----
#pragma unroll
  for (int t = 0; t < 8; ++t) {
#pragma unroll
    for (int v = 0; v < 8; ++v) {
      int row = v + 8 * h;
      int col = t * 16 + r;
      out[((size_t)b * S_ + s0 + row) * D_ + col] = o[t][v] / lrow[v];
    }
  }
}

extern "C" void kernel_launch(void* const* d_in, const int* in_sizes, int n_in,
                              void* d_out, int out_size, void* d_ws,
                              size_t ws_size, hipStream_t stream) {
  const float* x  = (const float*)d_in[0];
  const float* Wq = (const float*)d_in[1];
  const float* bq = (const float*)d_in[2];
  const float* Wk = (const float*)d_in[3];
  const float* bk = (const float*)d_in[4];
  const float* Wv = (const float*)d_in[5];
  const float* bv = (const float*)d_in[6];

  // Workspace: Q,K row-major bf16; V transposed bf16; transposed bf16 weights.
  size_t n = (size_t)B_ * S_ * D_;
  __bf16* Qbf = (__bf16*)d_ws;
  __bf16* Kbf = Qbf + n;
  __bf16* Vt  = Kbf + n;
  __bf16* Wt  = Vt + n;  // 3*128*128 bf16 = 96 KB

  prep_weights<<<dim3(D_, 3), dim3(D_), 0, stream>>>(Wq, Wk, Wv, Wt);

  dim3 grid(S_ / 16, B_);
  dim3 blk(32);
  qkv_pe_proj<<<grid, blk, 0, stream>>>(x, Wt, bq, bk, bv, Qbf, Kbf, Vt);
  flash_attn<<<grid, blk, 0, stream>>>(Qbf, Kbf, Vt, (float*)d_out);
}
